// HybridGAT_LSTM_90056874262835
// MI455X (gfx1250) — matrix operand order repaired and checked by reading, not compiled, not run on hardware
//
#include <hip/hip_runtime.h>
#include <math.h>

// ---------------------------------------------------------------------------
// HybridGAT_LSTM for MI455X (gfx1250).
// GEMM-shaped work -> v_wmma_f32_16x16x32_f16, wave32, one wave = 16x64 strip
// (4 N-tiles, A fragment reused across 4 WMMAs). All B (weight) operands are
// pre-swizzled once into the WMMA B-fragment layout so every operand load is
// a contiguous 32B (2x global_load_b128) access. f32 accumulate throughout.
// ---------------------------------------------------------------------------

typedef _Float16 h16;
typedef __attribute__((ext_vector_type(16))) _Float16 v16h;
typedef __attribute__((ext_vector_type(8)))  _Float16 v8h;
typedef __attribute__((ext_vector_type(8)))  float    v8f;

#define BB   2
#define TT   8
#define NN   4000
#define FIN  32
#define HH   4
#define CC   64
#define HLL  128
#define FHH  128
#define INST (BB*TT)     // 16 graph instances
#define ROWS (INST*NN)   // 64000
#define RB   (BB*NN)     // 8000  (LSTM batch)
#define GD   (HH*CC)     // 256
#define ZG   (4*HLL)     // 512  (LSTM gates)
#define KCAT 192         // C+HL (lstm in) and HL+C (fusion in)

static inline int cdiv(int a, int b) { return (a + b - 1) / b; }

// ------------------------------- WMMA GEMM ---------------------------------
// C[M,N] (f32) = A[M,K] (f16 row-major) * B (f16, pre-swizzled) [+bias][relu]
// Packed B layout: Bp[((tn*(K/32) + kb)*32 + lane)*16 + i]
//   lane = hi*16+lo : col = tn*16+lo ; K-row = kb*32 + hi*16 + i   (i in 0..15)
// One wave computes a 16x64 strip = 4 consecutive N-tiles.
// Requires M%16==0, N%64==0, K%32==0.
__device__ inline void store_tile(v8f acc, float* __restrict__ C, int N,
                                  int r0, int col, float bv, int relu) {
#pragma unroll
    for (int v = 0; v < 8; ++v) {
        float val = acc[v] + bv;
        if (relu) val = fmaxf(val, 0.0f);
        C[(size_t)(r0 + v) * N + col] = val;
    }
}

__global__ void k_gemm_wmma(const h16* __restrict__ A, const h16* __restrict__ Bp,
                            const float* __restrict__ bias, float* __restrict__ C,
                            int M, int N, int K, int relu)
{
    int wid     = blockIdx.x * (blockDim.x >> 5) + (threadIdx.x >> 5);
    int tilesN4 = N >> 6;                       // strips of 64 columns
    int tiles   = (M >> 4) * tilesN4;
    if (wid >= tiles) return;                   // wave-uniform; EXEC stays full
    int tm = wid / tilesN4, tg = wid % tilesN4;
    int m0 = tm << 4;
    int lane = threadIdx.x & 31;
    int hi = lane >> 4, lo = lane & 15;
    int kblocks = K >> 5;

    v8f acc0 = {}, acc1 = {}, acc2 = {}, acc3 = {};
    // packed-B base for tile (tg*4 + j, kb): offset ((tn*kblocks+kb)<<9) + lane*16
    const h16* Bl = Bp + (size_t)lane * 16;
    size_t tstride = (size_t)kblocks << 9;      // halves per N-tile block

    for (int kb = 0; kb < kblocks; ++kb) {
        // A 16x32 fragment: lane(lo,hi): row=m0+lo; elems 0..15 -> K=kb*32+hi*16+i.
        // Two contiguous 16B loads.
        const h16* Ar = A + (size_t)(m0 + lo) * K + (kb << 5) + hi * 16;
        v8h a0 = *(const v8h*)(Ar);
        v8h a1 = *(const v8h*)(Ar + 8);
        v16h a;
#pragma unroll
        for (int i = 0; i < 8; ++i) { a[i] = a0[i]; a[8 + i] = a1[i]; }

        const h16* B0 = Bl + ((size_t)(tg * 4) * kblocks + kb) * 512;
        v16h b0 = *(const v16h*)(B0);
        v16h b1 = *(const v16h*)(B0 + tstride);
        v16h b2 = *(const v16h*)(B0 + 2 * tstride);
        v16h b3 = *(const v16h*)(B0 + 3 * tstride);

        acc0 = __builtin_amdgcn_wmma_f32_16x16x32_f16(false, a, false, b0, (short)0, acc0, false, false);
        acc1 = __builtin_amdgcn_wmma_f32_16x16x32_f16(false, a, false, b1, (short)0, acc1, false, false);
        acc2 = __builtin_amdgcn_wmma_f32_16x16x32_f16(false, a, false, b2, (short)0, acc2, false, false);
        acc3 = __builtin_amdgcn_wmma_f32_16x16x32_f16(false, a, false, b3, (short)0, acc3, false, false);
    }
    // C/D layout: lane(lo,hi): col; elem v -> row = m0 + hi*8 + v.
    int r0 = m0 + hi * 8;
    int c0 = (tg * 4) * 16 + lo;
    store_tile(acc0, C, N, r0, c0,      bias ? bias[c0]      : 0.0f, relu);
    store_tile(acc1, C, N, r0, c0 + 16, bias ? bias[c0 + 16] : 0.0f, relu);
    store_tile(acc2, C, N, r0, c0 + 32, bias ? bias[c0 + 32] : 0.0f, relu);
    store_tile(acc3, C, N, r0, c0 + 48, bias ? bias[c0 + 48] : 0.0f, relu);
}

// --------------------------- weight pre-swizzle ----------------------------
// Pack f32 [K,N] row-major weight into the WMMA B-fragment layout (f16).
__global__ void k_pack_b(const float* __restrict__ src, h16* __restrict__ dst,
                         int K, int N) {
    int i = blockIdx.x * blockDim.x + threadIdx.x;       // K*N
    if (i >= K * N) return;
    int ii   = i & 15;
    int lane = (i >> 4) & 31;
    int kblocks = K >> 5;
    int kb = (i >> 9) % kblocks;
    int tn = (i >> 9) / kblocks;
    int hi = lane >> 4, lo = lane & 15;
    int krow = (kb << 5) + (hi << 4) + ii;
    int col  = (tn << 4) + lo;
    dst[i] = (h16)src[(size_t)krow * N + col];
}
// Pack the LSTM concat weight Wcat[k,j] = (k<64 ? W_ih[j,k] : W_hh[j,k-64]),
// K=192, N=512, directly into the swizzled layout.
__global__ void k_pack_wcat(const float* __restrict__ Wih, const float* __restrict__ Whh,
                            h16* __restrict__ dst) {
    int i = blockIdx.x * blockDim.x + threadIdx.x;       // KCAT*ZG
    if (i >= KCAT * ZG) return;
    int ii   = i & 15;
    int lane = (i >> 4) & 31;
    const int kblocks = KCAT >> 5;                       // 6
    int kb = (i >> 9) % kblocks;
    int tn = (i >> 9) / kblocks;
    int hi = lane >> 4, lo = lane & 15;
    int k   = (kb << 5) + (hi << 4) + ii;
    int j   = (tn << 4) + lo;
    float v = (k < CC) ? Wih[(size_t)j * CC + k] : Whh[(size_t)j * HLL + (k - CC)];
    dst[i] = (h16)v;
}

// ----------------------------- small helpers -------------------------------
__global__ void k_f32_to_f16(const float* __restrict__ s, h16* __restrict__ d, int n) {
    int i = blockIdx.x * blockDim.x + threadIdx.x;
    if (i < n) d[i] = (h16)s[i];
}
__global__ void k_fill(float* p, float v, int n) {
    int i = blockIdx.x * blockDim.x + threadIdx.x;
    if (i < n) p[i] = v;
}
__global__ void k_vec_add(const float* a, const float* b, float* o, int n) {
    int i = blockIdx.x * blockDim.x + threadIdx.x;
    if (i < n) o[i] = a[i] + b[i];
}

// --------------------------- GAT attention ops -----------------------------
__global__ void k_node_attn(const float* __restrict__ hbuf,
                            const float* __restrict__ asrc, const float* __restrict__ adst,
                            float* __restrict__ als, float* __restrict__ ald) {
    int i = blockIdx.x * blockDim.x + threadIdx.x;      // ROWS*HH
    if (i >= ROWS * HH) return;
    int row = i / HH, h = i % HH;
    const float* hp = hbuf + (size_t)row * GD + h * CC;
    float s = 0.f, d = 0.f;
    for (int c = 0; c < CC; ++c) { s += hp[c] * asrc[h * CC + c]; d += hp[c] * adst[h * CC + c]; }
    als[i] = s; ald[i] = d;
}

__device__ inline void edge_sd(const int* ei, int E, int e, int& s, int& d) {
    if (e < E) { s = ei[e]; d = ei[E + e]; } else { s = e - E; d = e - E; }  // self-loops
}
__device__ inline float atomicMaxF(float* addr, float val) {
    int* ia = (int*)addr;
    int old = __float_as_int(*addr);
    while (__int_as_float(old) < val) {
        int assumed = old;
        old = atomicCAS(ia, assumed, __float_as_int(val));
        if (old == assumed) break;
    }
    return __int_as_float(old);
}
__device__ inline float leaky(float x) { return x > 0.f ? x : 0.2f * x; }

__global__ void k_edge_max(const int* __restrict__ ei, int E,
                           const float* __restrict__ als, const float* __restrict__ ald,
                           float* __restrict__ mmax) {
    int Epl = E + NN;
    int i = blockIdx.x * blockDim.x + threadIdx.x;      // INST*Epl
    if (i >= INST * Epl) return;
    int inst = i / Epl, e = i % Epl;
    int s, d; edge_sd(ei, E, e, s, d);
    int rs = inst * NN + s, rd = inst * NN + d;
    for (int h = 0; h < HH; ++h)
        atomicMaxF(&mmax[rd * HH + h], leaky(als[rs * HH + h] + ald[rd * HH + h]));
}
__global__ void k_edge_expsum(const int* __restrict__ ei, int E,
                              const float* __restrict__ als, const float* __restrict__ ald,
                              const float* __restrict__ mmax,
                              float* __restrict__ eexp, float* __restrict__ denom) {
    int Epl = E + NN;
    int i = blockIdx.x * blockDim.x + threadIdx.x;      // INST*Epl
    if (i >= INST * Epl) return;
    int inst = i / Epl, e = i % Epl;
    int s, d; edge_sd(ei, E, e, s, d);
    int rs = inst * NN + s, rd = inst * NN + d;
    for (int h = 0; h < HH; ++h) {
        float x  = leaky(als[rs * HH + h] + ald[rd * HH + h]);
        float ev = expf(x - mmax[rd * HH + h]);
        eexp[(size_t)i * HH + h] = ev;
        atomicAdd(&denom[rd * HH + h], ev);
    }
}
__global__ void k_edge_scatter(const int* __restrict__ ei, int E,
                               const float* __restrict__ eexp,
                               const float* __restrict__ hbuf, float* __restrict__ gout) {
    int Epl = E + NN;
    int i = blockIdx.x * blockDim.x + threadIdx.x;      // INST*Epl*HH
    if (i >= INST * Epl * HH) return;
    int h = i % HH; int ie = i / HH;
    int inst = ie / Epl, e = ie % Epl;
    int s, d; edge_sd(ei, E, e, s, d);
    int rs = inst * NN + s, rd = inst * NN + d;
    float w = eexp[i];
    const float* hp = hbuf + (size_t)rs * GD + h * CC;
    float*       gp = gout + (size_t)rd * GD + h * CC;
    for (int c = 0; c < CC; ++c) atomicAdd(&gp[c], w * hp[c]);
}
__global__ void k_gat_final(const float* __restrict__ gout, const float* __restrict__ denom,
                            const float* __restrict__ bias, float* __restrict__ out) {
    int i = blockIdx.x * blockDim.x + threadIdx.x;      // ROWS*CC
    if (i >= ROWS * CC) return;
    int row = i / CC, c = i % CC;
    float s = 0.f;
    for (int h = 0; h < HH; ++h)
        s += gout[(size_t)row * GD + h * CC + c] / denom[row * HH + h];
    out[i] = fmaxf(s * (1.0f / HH) + bias[c], 0.0f);    // mean over heads + b, relu
}

// ------------------------------- LSTM / head -------------------------------
__global__ void k_build_xcat(const float* __restrict__ g2, const float* __restrict__ h,
                             h16* __restrict__ xcat, int t) {
    int i = blockIdx.x * blockDim.x + threadIdx.x;      // RB*KCAT
    if (i >= RB * KCAT) return;
    int r = i / KCAT, k = i % KCAT;
    int b = r / NN, n = r % NN;
    float v = (k < CC) ? g2[((size_t)(b * TT + t) * NN + n) * CC + k]
                       : h[(size_t)r * HLL + (k - CC)];
    xcat[i] = (h16)v;
}
__global__ void k_lstm_cell(const float* __restrict__ z, float* __restrict__ h,
                            float* __restrict__ c) {
    int i = blockIdx.x * blockDim.x + threadIdx.x;      // RB*HLL
    if (i >= RB * HLL) return;
    int r = i / HLL, j = i % HLL;
    const float* zr = z + (size_t)r * ZG;
    float ig = 1.f / (1.f + expf(-zr[j]));
    float fg = 1.f / (1.f + expf(-zr[HLL + j]));
    float gg = tanhf(zr[2 * HLL + j]);
    float og = 1.f / (1.f + expf(-zr[3 * HLL + j]));
    float cn = fg * c[i] + ig * gg;
    c[i] = cn;
    h[i] = og * tanhf(cn);
}
__global__ void k_build_fusecat(const float* __restrict__ h, const float* __restrict__ g2,
                                h16* __restrict__ xcat) {
    int i = blockIdx.x * blockDim.x + threadIdx.x;      // RB*KCAT
    if (i >= RB * KCAT) return;
    int r = i / KCAT, k = i % KCAT;
    int b = r / NN, n = r % NN;
    float v = (k < HLL) ? h[(size_t)r * HLL + k]
                        : g2[((size_t)(b * TT + (TT - 1)) * NN + n) * CC + (k - HLL)];
    xcat[i] = (h16)v;
}
__global__ void k_out_head(const float* __restrict__ fused, const float* __restrict__ Wo,
                           const float* __restrict__ bo, float* __restrict__ out) {
    int r = blockIdx.x * blockDim.x + threadIdx.x;      // RB
    if (r >= RB) return;
    float s = bo[0];
    for (int j = 0; j < FHH; ++j) s += fused[(size_t)r * FHH + j] * Wo[j];
    out[r] = s;                                          // d_out layout = [B,N]
}

// --------------------------------- launch ----------------------------------
extern "C" void kernel_launch(void* const* d_in, const int* in_sizes, int n_in,
                              void* d_out, int out_size, void* d_ws, size_t ws_size,
                              hipStream_t stream) {
    const float* x_seq = (const float*)d_in[0];
    const float* W1    = (const float*)d_in[1];
    const float* asrc1 = (const float*)d_in[2];
    const float* adst1 = (const float*)d_in[3];
    const float* b1    = (const float*)d_in[4];
    const float* W2    = (const float*)d_in[5];
    const float* asrc2 = (const float*)d_in[6];
    const float* adst2 = (const float*)d_in[7];
    const float* b2    = (const float*)d_in[8];
    const float* W_ih  = (const float*)d_in[9];
    const float* W_hh  = (const float*)d_in[10];
    const float* b_ih  = (const float*)d_in[11];
    const float* b_hh  = (const float*)d_in[12];
    const float* Wf    = (const float*)d_in[13];
    const float* bf    = (const float*)d_in[14];
    const float* Wo    = (const float*)d_in[15];
    const float* bo    = (const float*)d_in[16];
    const int*   ei    = (const int*)d_in[17];
    const int    E     = in_sizes[17] / 2;              // 40000
    const int    Epl   = E + NN;                        // + self loops

    // ---- workspace carve-up (aligned 256B) ----
    size_t off = 0;
    auto alloc = [&](size_t bytes) -> void* {
        void* p = (char*)d_ws + off;
        off += (bytes + 255) & ~(size_t)255;
        return p;
    };
    h16*   W1p   = (h16*)  alloc((size_t)FIN * GD * sizeof(h16));
    h16*   W2p   = (h16*)  alloc((size_t)CC * GD * sizeof(h16));
    h16*   Wcatp = (h16*)  alloc((size_t)KCAT * ZG * sizeof(h16));
    h16*   Wfp   = (h16*)  alloc((size_t)KCAT * FHH * sizeof(h16));
    float* blstm = (float*)alloc((size_t)ZG * sizeof(float));
    h16*   xh    = (h16*)  alloc((size_t)ROWS * FIN * sizeof(h16));
    float* hbuf  = (float*)alloc((size_t)ROWS * GD * sizeof(float));
    float* als   = (float*)alloc((size_t)ROWS * HH * sizeof(float));
    float* ald   = (float*)alloc((size_t)ROWS * HH * sizeof(float));
    float* mmax  = (float*)alloc((size_t)ROWS * HH * sizeof(float));
    float* denom = (float*)alloc((size_t)ROWS * HH * sizeof(float));
    float* eexp  = (float*)alloc((size_t)INST * Epl * HH * sizeof(float));
    float* gout  = (float*)alloc((size_t)ROWS * GD * sizeof(float));
    float* g1    = (float*)alloc((size_t)ROWS * CC * sizeof(float));
    h16*   g1h   = (h16*)  alloc((size_t)ROWS * CC * sizeof(h16));
    float* g2    = (float*)alloc((size_t)ROWS * CC * sizeof(float));
    float* hst   = (float*)alloc((size_t)RB * HLL * sizeof(float));
    float* cst   = (float*)alloc((size_t)RB * HLL * sizeof(float));
    h16*   xcat  = (h16*)  alloc((size_t)RB * KCAT * sizeof(h16));
    float* zbuf  = (float*)alloc((size_t)RB * ZG * sizeof(float));
    float* fused = (float*)alloc((size_t)RB * FHH * sizeof(float));
    (void)ws_size; (void)n_in; (void)out_size;

    auto gemm = [&](const h16* A, const h16* Bp, const float* bias, float* Cm,
                    int M, int N, int K, int relu) {
        int waves = (M / 16) * (N / 64);                 // one wave = 16x64 strip
        k_gemm_wmma<<<cdiv(waves, 8), 256, 0, stream>>>(A, Bp, bias, Cm, M, N, K, relu);
    };

    // ---- weight prep: convert + swizzle into WMMA B-fragment layout ----
    k_pack_b<<<cdiv(FIN * GD, 256), 256, 0, stream>>>(W1, W1p, FIN, GD);
    k_pack_b<<<cdiv(CC * GD, 256), 256, 0, stream>>>(W2, W2p, CC, GD);
    k_pack_wcat<<<cdiv(KCAT * ZG, 256), 256, 0, stream>>>(W_ih, W_hh, Wcatp);
    k_pack_b<<<cdiv(KCAT * FHH, 256), 256, 0, stream>>>(Wf, Wfp, KCAT, FHH);
    k_vec_add<<<cdiv(ZG, 256), 256, 0, stream>>>(b_ih, b_hh, blstm, ZG);
    k_f32_to_f16<<<cdiv(ROWS * FIN, 256), 256, 0, stream>>>(x_seq, xh, ROWS * FIN);

    // ---- one GAT layer (shared routine) ----
    auto gat_layer = [&](const h16* Ah, int K, const h16* Wp,
                         const float* as, const float* ad, const float* bias, float* outg) {
        gemm(Ah, Wp, nullptr, hbuf, ROWS, GD, K, 0);    // h = x@W -> [ROWS, H*C]
        k_node_attn<<<cdiv(ROWS * HH, 256), 256, 0, stream>>>(hbuf, as, ad, als, ald);
        k_fill<<<cdiv(ROWS * HH, 256), 256, 0, stream>>>(mmax, -3.0e38f, ROWS * HH);
        k_fill<<<cdiv(ROWS * HH, 256), 256, 0, stream>>>(denom, 0.0f, ROWS * HH);
        k_fill<<<cdiv(ROWS * GD, 256), 256, 0, stream>>>(gout, 0.0f, ROWS * GD);
        k_edge_max<<<cdiv(INST * Epl, 256), 256, 0, stream>>>(ei, E, als, ald, mmax);
        k_edge_expsum<<<cdiv(INST * Epl, 256), 256, 0, stream>>>(ei, E, als, ald, mmax, eexp, denom);
        k_edge_scatter<<<cdiv(INST * Epl * HH, 256), 256, 0, stream>>>(ei, E, eexp, hbuf, gout);
        k_gat_final<<<cdiv(ROWS * CC, 256), 256, 0, stream>>>(gout, denom, bias, outg);
    };

    // ---- GAT x2 ----
    gat_layer(xh, FIN, W1p, asrc1, adst1, b1, g1);
    k_f32_to_f16<<<cdiv(ROWS * CC, 256), 256, 0, stream>>>(g1, g1h, ROWS * CC);
    gat_layer(g1h, CC, W2p, asrc2, adst2, b2, g2);

    // ---- LSTM over T=8, batch B*N=8000 ----
    k_fill<<<cdiv(RB * HLL, 256), 256, 0, stream>>>(hst, 0.0f, RB * HLL);
    k_fill<<<cdiv(RB * HLL, 256), 256, 0, stream>>>(cst, 0.0f, RB * HLL);
    for (int t = 0; t < TT; ++t) {
        k_build_xcat<<<cdiv(RB * KCAT, 256), 256, 0, stream>>>(g2, hst, xcat, t);
        gemm(xcat, Wcatp, blstm, zbuf, RB, ZG, KCAT, 0); // z = [x|h]@Wcat + (b_ih+b_hh)
        k_lstm_cell<<<cdiv(RB * HLL, 256), 256, 0, stream>>>(zbuf, hst, cst);
    }

    // ---- fusion + head ----
    k_build_fusecat<<<cdiv(RB * KCAT, 256), 256, 0, stream>>>(hst, g2, xcat);
    gemm(xcat, Wfp, bf, fused, RB, FHH, KCAT, 1);        // relu(cat @ Wf + bf)
    k_out_head<<<cdiv(RB, 256), 256, 0, stream>>>(fused, Wo, bo, (float*)d_out);
}
